// GraphProp_14628658610878
// MI455X (gfx1250) — matrix operand order repaired and checked by reading, not compiled, not run on hardware
//
#include <hip/hip_runtime.h>

// ---------------------------------------------------------------------------
// GraphProp on MI455X (gfx1250, wave32, WMMA).
//  - Message GEMM [E,129]x[129,64] and GRU GEMMs [N,64]x[64,192] run on
//    v_wmma_f32_16x16x32_f16 (f16 inputs, f32 accumulate).
//  - x (25.6MB) and agg (25.6MB) are L2-resident (192MB L2), so edge gathers
//    and float-atomic segment-sum stay on-chip; HBM sees mostly edge_index.
// ---------------------------------------------------------------------------

typedef __attribute__((ext_vector_type(16))) _Float16 v16h;
typedef __attribute__((ext_vector_type(8)))  float    v8f;
typedef __attribute__((ext_vector_type(4)))  unsigned v4u;
typedef __attribute__((ext_vector_type(4)))  float    v4f;

#define NNODES 100000
#define NEDGES 1250000
#define HDIM   64

union ABv { v4u u[2]; v16h v; };   // 32 bytes: one WMMA A/B fragment per lane

static __device__ __forceinline__ unsigned pkh(float a, float b) {
  union { _Float16 h[2]; unsigned u; } x;
  x.h[0] = (_Float16)a; x.h[1] = (_Float16)b;
  return x.u;
}
static __device__ __forceinline__ unsigned short f2h(float a) {
  union { _Float16 h; unsigned short u; } x;
  x.h = (_Float16)a; return x.u;
}
static __device__ __forceinline__ float sigm(float x) {
  return 1.0f / (1.0f + __expf(-x));
}

// ---------------------------------------------------------------------------
// Zero the aggregation buffer (runs every round; ws is not re-poisoned).
__global__ void zero_kernel(v4f* __restrict__ p, int n4) {
  int i = blockIdx.x * blockDim.x + threadIdx.x;
  if (i < n4) { v4f z = {0.f, 0.f, 0.f, 0.f}; p[i] = z; }
}

// ---------------------------------------------------------------------------
// Pack msg_W [T,64,129] (f32) -> f16 B fragments for 16x16x32 WMMA.
// B[k][n] = W[n][k].  Fragment f = (t*4 + nt)*4 + kk, 32 lanes x 16 halves.
// Lane layout: lanes 0-15 hold N=lane, K = kk*32 + 0..15 (consecutive);
//              lanes 16-31 hold N=lane-16, K = kk*32 + 16..31.
__global__ void pack_msg_kernel(const float* __restrict__ W,
                                unsigned short* __restrict__ Bm) {
  const int tid = blockIdx.x * blockDim.x + threadIdx.x;
  if (tid >= 2 * 4 * 4 * 32) return;
  const int lane = tid & 31;
  const int f = tid >> 5;
  const int kk = f & 3, nt = (f >> 2) & 3, t = f >> 4;
  const int kbase = kk * 32 + ((lane < 16) ? 0 : 16);
  const int n = nt * 16 + (lane & 15);
  unsigned short* out = Bm + (size_t)tid * 16;
#pragma unroll
  for (int p = 0; p < 16; ++p)
    out[p] = f2h(W[(size_t)(t * HDIM + n) * 129 + (kbase + p)]);
}

// Pack gru_Wih / gru_Whh [T,192,64] -> f16 B fragments.
// Fragment f = (t*12 + nt)*2 + kk (nt over 192 cols, kk over K=64).
__global__ void pack_gru_kernel(const float* __restrict__ Wih,
                                const float* __restrict__ Whh,
                                unsigned short* __restrict__ Bi,
                                unsigned short* __restrict__ Bh) {
  const int tid = blockIdx.x * blockDim.x + threadIdx.x;
  if (tid >= 2 * 12 * 2 * 32) return;
  const int lane = tid & 31;
  const int f = tid >> 5;
  const int kk = f & 1;
  const int tn = f >> 1;
  const int nt = tn % 12, t = tn / 12;
  const int kbase = kk * 32 + ((lane < 16) ? 0 : 16);
  const int n = nt * 16 + (lane & 15);
#pragma unroll
  for (int p = 0; p < 16; ++p) {
    const int k = kbase + p;
    Bi[(size_t)tid * 16 + p] = f2h(Wih[(size_t)(t * 192 + n) * HDIM + k]);
    Bh[(size_t)tid * 16 + p] = f2h(Whh[(size_t)(t * 192 + n) * HDIM + k]);
  }
}

// ---------------------------------------------------------------------------
// Per-edge messages + atomic segment-sum. One wave = one 16-edge tile.
// A tile: 16 edges x 128 K (K 0-63 = dest = x[col], K 64-127 = src = x[row]).
// Lane l: eL = l&15, part = l>>4 gathers one 64-float feature row -> f16 LDS.
__global__ __launch_bounds__(256) void msg_agg_kernel(
    const float* __restrict__ x, const long long* __restrict__ ei,
    const float* __restrict__ eattr,
    const float* __restrict__ msgW, const float* __restrict__ msgb,
    const v4u* __restrict__ Bm, float* __restrict__ agg, int t) {
  __shared__ unsigned shA[8][16][68];   // 64 data dwords + 4 pad per row
  __shared__ int      shRow[8][16];
  __shared__ float    shAttr[8][16];

  const int lane = threadIdx.x & 31;
  const int w    = threadIdx.x >> 5;
  const int tile = blockIdx.x * 8 + w;
  const bool active = (tile * 16) < NEDGES;   // wave-uniform
  const int e0 = tile * 16;
  const int eL = lane & 15, part = lane >> 4;

  if (active) {
    const int e = e0 + eL;
    // dest (K 0-63) = x[col] = ei[E+e]; src (K 64-127) = x[row] = ei[e]
    const long long node = part ? ei[e] : ei[(long long)NEDGES + e];
    if (part == 0) { shRow[w][eL] = (int)ei[e]; shAttr[w][eL] = eattr[e]; }
    const v4f* xv = (const v4f*)x;
    unsigned* dst = &shA[w][eL][part * 32];
#pragma unroll
    for (int i = 0; i < 8; ++i) {
      v4f a = xv[node * 16 + 2 * i];
      v4f b = xv[node * 16 + 2 * i + 1];
      v4u q; q.x = pkh(a.x, a.y); q.y = pkh(a.z, a.w);
             q.z = pkh(b.x, b.y); q.w = pkh(b.z, b.w);
      *(v4u*)(dst + 4 * i) = q;
    }
  }
  __syncthreads();
  if (!active) return;

  // A fragment per k-step: row M = lane&15; halves kbase..+15 (contiguous)
  const unsigned* rp = shA[w][lane & 15];
  const int aoff = (lane < 16) ? 0 : 8;   // dwords
  v8f acc[4] = {};
#pragma unroll
  for (int kk = 0; kk < 4; ++kk) {
    ABv a;
    a.u[0] = *(const v4u*)(rp + kk * 16 + aoff);
    a.u[1] = *(const v4u*)(rp + kk * 16 + aoff + 4);
#pragma unroll
    for (int nt = 0; nt < 4; ++nt) {
      ABv b;
      const int f = (t * 4 + nt) * 4 + kk;
      b.u[0] = Bm[(f * 32 + lane) * 2 + 0];
      b.u[1] = Bm[(f * 32 + lane) * 2 + 1];
      acc[nt] = __builtin_amdgcn_wmma_f32_16x16x32_f16(
          false, a.v, false, b.v, (short)0, acc[nt], false, false);
    }
  }

  // C layout: lane gives N (lane&15), VGPR j gives M = j + 8*(lane>=16).
  const int nl = lane & 15;
  const int mh = (lane >> 4) * 8;
#pragma unroll
  for (int nt = 0; nt < 4; ++nt) {
    const int n = nt * 16 + nl;
    const float bias = msgb[t * HDIM + n];
    const float wat  = msgW[(size_t)(t * HDIM + n) * 129 + 128];
#pragma unroll
    for (int j = 0; j < 8; ++j) {
      const int m = mh + j;
      const float v = acc[nt][j] + bias + shAttr[w][m] * wat;
      atomicAdd(&agg[(long long)shRow[w][m] * HDIM + n], v);
    }
  }
}

// ---------------------------------------------------------------------------
// GRUCell: one wave = 16 nodes. gi = agg@Wih.T+bih, gh = h@Whh.T+bhh via
// WMMA (K=64 -> 2 k-steps), gates r/z/n per 16-channel block, then pointwise.
__global__ __launch_bounds__(256) void gru_kernel(
    const float* __restrict__ agg, const float* __restrict__ xin,
    float* __restrict__ xout,
    const v4u* __restrict__ Bih, const v4u* __restrict__ Bhh,
    const float* __restrict__ bih, const float* __restrict__ bhh, int t) {
  __shared__ unsigned shG[8][16][36];   // agg tile f16 (32 dwords + 4 pad)
  __shared__ unsigned shH[8][16][36];   // h tile f16

  const int lane = threadIdx.x & 31;
  const int w    = threadIdx.x >> 5;
  const int tile = blockIdx.x * 8 + w;
  const bool active = (tile * 16) < NNODES;
  const int n0 = tile * 16;
  const int nl = lane & 15, part = lane >> 4;

  if (active) {
    const long long node = n0 + nl;
    const v4f* gv = (const v4f*)agg;
    const v4f* hv = (const v4f*)xin;
    unsigned* dg = &shG[w][nl][part * 16];
    unsigned* dh = &shH[w][nl][part * 16];
#pragma unroll
    for (int i = 0; i < 4; ++i) {
      v4f a = gv[node * 16 + part * 8 + 2 * i];
      v4f b = gv[node * 16 + part * 8 + 2 * i + 1];
      v4u q; q.x = pkh(a.x, a.y); q.y = pkh(a.z, a.w);
             q.z = pkh(b.x, b.y); q.w = pkh(b.z, b.w);
      *(v4u*)(dg + 4 * i) = q;
      a = hv[node * 16 + part * 8 + 2 * i];
      b = hv[node * 16 + part * 8 + 2 * i + 1];
      v4u r; r.x = pkh(a.x, a.y); r.y = pkh(a.z, a.w);
             r.z = pkh(b.x, b.y); r.w = pkh(b.z, b.w);
      *(v4u*)(dh + 4 * i) = r;
    }
  }
  __syncthreads();
  if (!active) return;

  const unsigned* rg = shG[w][nl];
  const unsigned* rh = shH[w][nl];
  const int aoff = (lane < 16) ? 0 : 8;
  const int mh = (lane >> 4) * 8;

  for (int nb = 0; nb < 4; ++nb) {       // 16 output channels per iteration
    v8f ir = {}, iz = {}, in_ = {}, hr = {}, hz = {}, hn = {};
#pragma unroll
    for (int kk = 0; kk < 2; ++kk) {
      ABv ag, ax;
      ag.u[0] = *(const v4u*)(rg + kk * 16 + aoff);
      ag.u[1] = *(const v4u*)(rg + kk * 16 + aoff + 4);
      ax.u[0] = *(const v4u*)(rh + kk * 16 + aoff);
      ax.u[1] = *(const v4u*)(rh + kk * 16 + aoff + 4);
      const int fr = ((t * 12) + nb) * 2 + kk;        // r-gate cols nb*16
      const int fz = ((t * 12) + 4 + nb) * 2 + kk;    // z-gate cols 64+nb*16
      const int fn = ((t * 12) + 8 + nb) * 2 + kk;    // n-gate cols 128+nb*16
      ABv b;
      b.u[0] = Bih[(fr * 32 + lane) * 2]; b.u[1] = Bih[(fr * 32 + lane) * 2 + 1];
      ir = __builtin_amdgcn_wmma_f32_16x16x32_f16(false, ag.v, false, b.v, (short)0, ir, false, false);
      b.u[0] = Bhh[(fr * 32 + lane) * 2]; b.u[1] = Bhh[(fr * 32 + lane) * 2 + 1];
      hr = __builtin_amdgcn_wmma_f32_16x16x32_f16(false, ax.v, false, b.v, (short)0, hr, false, false);
      b.u[0] = Bih[(fz * 32 + lane) * 2]; b.u[1] = Bih[(fz * 32 + lane) * 2 + 1];
      iz = __builtin_amdgcn_wmma_f32_16x16x32_f16(false, ag.v, false, b.v, (short)0, iz, false, false);
      b.u[0] = Bhh[(fz * 32 + lane) * 2]; b.u[1] = Bhh[(fz * 32 + lane) * 2 + 1];
      hz = __builtin_amdgcn_wmma_f32_16x16x32_f16(false, ax.v, false, b.v, (short)0, hz, false, false);
      b.u[0] = Bih[(fn * 32 + lane) * 2]; b.u[1] = Bih[(fn * 32 + lane) * 2 + 1];
      in_ = __builtin_amdgcn_wmma_f32_16x16x32_f16(false, ag.v, false, b.v, (short)0, in_, false, false);
      b.u[0] = Bhh[(fn * 32 + lane) * 2]; b.u[1] = Bhh[(fn * 32 + lane) * 2 + 1];
      hn = __builtin_amdgcn_wmma_f32_16x16x32_f16(false, ax.v, false, b.v, (short)0, hn, false, false);
    }
    const int c = nb * 16 + nl;
    const float b_ir = bih[t * 192 + c],       b_hr = bhh[t * 192 + c];
    const float b_iz = bih[t * 192 + 64 + c],  b_hz = bhh[t * 192 + 64 + c];
    const float b_in = bih[t * 192 + 128 + c], b_hn = bhh[t * 192 + 128 + c];
#pragma unroll
    for (int j = 0; j < 8; ++j) {
      const long long nd = n0 + mh + j;
      const float h  = xin[nd * HDIM + c];
      const float r  = sigm(ir[j] + b_ir + hr[j] + b_hr);
      const float z  = sigm(iz[j] + b_iz + hz[j] + b_hz);
      const float nn = tanhf(in_[j] + b_in + r * (hn[j] + b_hn));
      xout[nd * HDIM + c] = (1.0f - z) * nn + z * h;
    }
  }
}

// ---------------------------------------------------------------------------
extern "C" void kernel_launch(void* const* d_in, const int* in_sizes, int n_in,
                              void* d_out, int out_size, void* d_ws, size_t ws_size,
                              hipStream_t stream) {
  const float*     x     = (const float*)d_in[0];
  const long long* ei    = (const long long*)d_in[1];
  const float*     eattr = (const float*)d_in[2];
  const float*     msgW  = (const float*)d_in[3];
  const float*     msgb  = (const float*)d_in[4];
  const float*     Wih   = (const float*)d_in[5];
  const float*     bih   = (const float*)d_in[6];
  const float*     Whh   = (const float*)d_in[7];
  const float*     bhh   = (const float*)d_in[8];
  float* out = (float*)d_out;

  char* ws = (char*)d_ws;
  float* agg = (float*)ws;                                     // 25,600,000 B
  unsigned short* Bm = (unsigned short*)(ws + 25600000);       //     32,768 B
  unsigned short* Bi = (unsigned short*)(ws + 25600000 + 32768);       // 49,152 B
  unsigned short* Bh = (unsigned short*)(ws + 25600000 + 32768 + 49152);

  pack_msg_kernel<<<4, 256, 0, stream>>>(msgW, Bm);
  pack_gru_kernel<<<6, 256, 0, stream>>>(Wih, Whh, Bi, Bh);

  const int msg_blocks = (NEDGES / 16 + 7) / 8;   // 9766
  const int gru_blocks = (NNODES / 16 + 7) / 8;   // 782

  for (int t = 0; t < 2; ++t) {
    zero_kernel<<<(NNODES * HDIM / 4 + 255) / 256, 256, 0, stream>>>(
        (v4f*)agg, NNODES * HDIM / 4);
    const float* xin = (t == 0) ? x : out;
    msg_agg_kernel<<<msg_blocks, 256, 0, stream>>>(
        xin, ei, eattr, msgW, msgb, (const v4u*)Bm, agg, t);
    gru_kernel<<<gru_blocks, 256, 0, stream>>>(
        agg, xin, out, (const v4u*)Bi, (const v4u*)Bh, bih, bhh, t);
  }
}
